// LinearAttention_17042430230962
// MI455X (gfx1250) — matrix operand order repaired and checked by reading, not compiled, not run on hardware
//
#include <hip/hip_runtime.h>
#include <hip/hip_bf16.h>

// ---------------------------------------------------------------------------
// CDNA5 (gfx1250) linear-attention pipeline, bf16 WMMA (16x16x32, f32 acc)
// + async global->LDS tile staging (ASYNCcnt path) when available
// ---------------------------------------------------------------------------

typedef __attribute__((ext_vector_type(16))) __bf16        v16bf;
typedef __attribute__((ext_vector_type(8)))  float         v8f;
typedef __attribute__((ext_vector_type(4)))  unsigned int  u32x4;
typedef __attribute__((ext_vector_type(8)))  unsigned int  u32x8;

#define HEADS 8
#define DHEAD 64
#define BATCH 8
#define SEQ   4096
#define CH    512
#define QKVF  1536   // 3*H*D
#define ROWS  (BATCH*SEQ)   // 32768

#if defined(__gfx1250__) && \
    __has_builtin(__builtin_amdgcn_global_load_async_to_lds_b128) && \
    __has_builtin(__builtin_amdgcn_s_wait_asynccnt)
#define HAVE_ASYNC 1
#else
#define HAVE_ASYNC 0
#endif

#if HAVE_ASYNC
// builtin signature (from hipcc diagnostic):
//   (v4i addrspace(1)* src, v4i addrspace(3)* dst, imm int offset, imm int cpol)
typedef int v4i __attribute__((vector_size(16)));
typedef __attribute__((address_space(1))) v4i* gv4i_p;
typedef __attribute__((address_space(3))) v4i* lv4i_p;

// one 16-byte per-lane async copy: global -> LDS, tracked by ASYNCcnt
__device__ __forceinline__ void async_copy16(const unsigned short* g,
                                             unsigned short* l) {
  __builtin_amdgcn_global_load_async_to_lds_b128((gv4i_p)g, (lv4i_p)l, 0, 0);
}
__device__ __forceinline__ void async_wait_all() {
  __builtin_amdgcn_s_wait_asynccnt(0);
}
#else
__device__ __forceinline__ void async_copy16(const unsigned short* g,
                                             unsigned short* l) {
  *(u32x4*)l = *(const u32x4*)g;   // synchronous fallback
}
__device__ __forceinline__ void async_wait_all() {}
#endif

__device__ __forceinline__ unsigned short f2bf(float f) {
  unsigned u = __builtin_bit_cast(unsigned, f);
  unsigned r = u + 0x7FFFu + ((u >> 16) & 1u);   // round-to-nearest-even
  return (unsigned short)(r >> 16);
}
__device__ __forceinline__ float bf2f(unsigned short h) {
  unsigned u = ((unsigned)h) << 16;
  return __builtin_bit_cast(float, u);
}

__device__ __forceinline__ v8f wmma_bf16(v16bf a, v16bf b, v8f c) {
  // (neg_a, A, neg_b, B, c_mod, C, reuse_a, reuse_b)
  return __builtin_amdgcn_wmma_f32_16x16x32_bf16(false, a, false, b, (short)0, c,
                                                 false, false);
}

// A fragment: 16x32 (MxK), row-major staging with 'ld' u16 elements per row.
// Lanes 0-15: row=lane, K 0..7 + 16..23 ; lanes 16-31: row=lane-16, K 8..15 + 24..31
__device__ __forceinline__ v16bf load_a_frag(const unsigned short* base, int ld,
                                             int row0, int k0) {
  int lane = threadIdx.x & 31;
  const unsigned short* p = base + (row0 + (lane & 15)) * ld + k0 + ((lane >> 4) << 3);
  u32x4 lo = *(const u32x4*)p;
  u32x4 hi = *(const u32x4*)(p + 16);
  u32x8 w;
  w[0]=lo[0]; w[1]=lo[1]; w[2]=lo[2]; w[3]=lo[3];
  w[4]=hi[0]; w[5]=hi[1]; w[6]=hi[2]; w[7]=hi[3];
  return __builtin_bit_cast(v16bf, w);
}

// B fragment: 32x16 (KxN) from K-transposed staging Bt[n][k] ('ld' per column row).
// Lanes 0-15: col=lane, K 0..7 + 16..23 ; lanes 16-31: col=lane-16, K 8..15 + 24..31
__device__ __forceinline__ v16bf load_b_frag(const unsigned short* base, int ld,
                                             int col0, int k0) {
  int lane = threadIdx.x & 31;
  const unsigned short* p = base + (col0 + (lane & 15)) * ld + k0 + ((lane >> 4) << 3);
  u32x4 lo = *(const u32x4*)p;
  u32x4 hi = *(const u32x4*)(p + 16);
  u32x8 w;
  w[0]=lo[0]; w[1]=lo[1]; w[2]=lo[2]; w[3]=lo[3];
  w[4]=hi[0]; w[5]=hi[1]; w[6]=hi[2]; w[7]=hi[3];
  return __builtin_bit_cast(v16bf, w);
}

// ---------------------------------------------------------------------------
// 0) fp32 -> bf16 conversion (x, w_qkv, w_out)
// ---------------------------------------------------------------------------
__global__ void k_f32_to_bf16(const float* __restrict__ src,
                              unsigned short* __restrict__ dst, int n) {
  int i = blockIdx.x * blockDim.x + threadIdx.x;
  if (i < n) dst[i] = f2bf(src[i]);
}

// ---------------------------------------------------------------------------
// 1) qkv = x @ w_qkv   (32768x512 @ 512x1536), 128x128 tiles, 8 waves
//    wave computes 64x32 = 4x2 WMMA tiles
// ---------------------------------------------------------------------------
__global__ __launch_bounds__(256) void k_gemm_qkv(
    const unsigned short* __restrict__ xb,     // [32768][512]
    const unsigned short* __restrict__ wb,     // [512][1536]
    unsigned short* __restrict__ qkvb) {       // [32768][1536]
  __shared__ alignas(16) unsigned short As[128 * 32];
  __shared__ alignas(16) unsigned short Bs[128 * 32];   // staged as [n][k]
  const int t = threadIdx.x;
  const int wave = t >> 5;
  const int row0 = blockIdx.x * 128;
  const int col0 = blockIdx.y * 128;
  const int wrow = (wave & 1) * 64;
  const int wcol = (wave >> 1) * 32;

  const v8f zero = {0.f, 0.f, 0.f, 0.f, 0.f, 0.f, 0.f, 0.f};
  v8f acc[4][2];
  for (int i = 0; i < 4; ++i)
    for (int j = 0; j < 2; ++j) acc[i][j] = zero;

  for (int kk = 0; kk < CH; kk += 32) {
    // stage A (128x32): two 16B async chunks per thread
    {
      int r0 = t >> 2, kc0 = (t & 3) << 3;
      async_copy16(xb + (size_t)(row0 + r0) * CH + kk + kc0, As + r0 * 32 + kc0);
      int c1 = t + 256;
      int r1 = c1 >> 2, kc1 = (c1 & 3) << 3;
      async_copy16(xb + (size_t)(row0 + r1) * CH + kk + kc1, As + r1 * 32 + kc1);
    }
    // stage B transposed: Bt[f][k] from wb[k][f] (coalesced global reads)
    for (int i = 0; i < 16; ++i) {
      int e = t + i * 256;             // 0..4095
      int k = e >> 7, f = e & 127;
      Bs[f * 32 + k] = wb[(size_t)(kk + k) * QKVF + col0 + f];
    }
    if (kk + 32 < CH)  // warm L2 for next A chunk
      __builtin_prefetch(xb + (size_t)(row0 + (t >> 2)) * CH + kk + 32, 0, 1);
    async_wait_all();
    __syncthreads();

    v16bf a[4], b[2];
    for (int i = 0; i < 4; ++i) a[i] = load_a_frag(As, 32, wrow + i * 16, 0);
    for (int j = 0; j < 2; ++j) b[j] = load_b_frag(Bs, 32, wcol + j * 16, 0);
    for (int i = 0; i < 4; ++i)
      for (int j = 0; j < 2; ++j) acc[i][j] = wmma_bf16(a[i], b[j], acc[i][j]);
    __syncthreads();
  }

  const int lane = t & 31;
  const int cidx = lane & 15, rbase = (lane >> 4) << 3;
  for (int i = 0; i < 4; ++i)
    for (int j = 0; j < 2; ++j) {
      int rg = row0 + wrow + i * 16 + rbase;
      int cg = col0 + wcol + j * 16 + cidx;
      for (int r = 0; r < 8; ++r)
        qkvb[(size_t)(rg + r) * QKVF + cg] = f2bf(acc[i][j][r]);
    }
}

// ---------------------------------------------------------------------------
// 2a) q softmax over D (per row), *1/sqrt(D); write qhat[b,h,n,d] (bf16)
// ---------------------------------------------------------------------------
__global__ void k_softmax_q(const unsigned short* __restrict__ qkvb,
                            unsigned short* __restrict__ qhat) {
  int idx = blockIdx.x * blockDim.x + threadIdx.x;   // (b*N+n)*H + h
  if (idx >= ROWS * HEADS) return;
  int h = idx & 7, bn = idx >> 3;
  int b = bn >> 12, n = bn & 4095;
  const unsigned short* src = qkvb + (size_t)bn * QKVF + h * DHEAD;
  float v[DHEAD];
  float mx = -3.4e38f;
  for (int d = 0; d < DHEAD; ++d) { v[d] = bf2f(src[d]); mx = fmaxf(mx, v[d]); }
  float s = 0.f;
  for (int d = 0; d < DHEAD; ++d) { v[d] = __expf(v[d] - mx); s += v[d]; }
  float inv = 0.125f / s;    // 1/sqrt(64) folded in
  unsigned short* dst = qhat + ((size_t)(b * HEADS + h) * SEQ + n) * DHEAD;
  for (int d = 0; d < DHEAD; ++d) dst[d] = f2bf(v[d] * inv);
}

// ---------------------------------------------------------------------------
// 2b) k softmax over N (one block per (b,h,d) column); write khat[b,h,d,n]
// ---------------------------------------------------------------------------
__global__ __launch_bounds__(256) void k_softmax_k(
    const unsigned short* __restrict__ qkvb,
    unsigned short* __restrict__ khat) {
  __shared__ float red[256];
  const int col = blockIdx.x;       // (b*H+h)*D + d, 0..4095
  const int b = col >> 9;
  const int hd = col & 511;
  const int t = threadIdx.x;
  const unsigned short* src = qkvb + (size_t)b * SEQ * QKVF + CH + hd;
  float v[16];
  float mx = -3.4e38f;
  for (int i = 0; i < 16; ++i) {
    int n = t + i * 256;
    v[i] = bf2f(src[(size_t)n * QKVF]);
    mx = fmaxf(mx, v[i]);
  }
  red[t] = mx; __syncthreads();
  for (int s = 128; s > 0; s >>= 1) {
    if (t < s) red[t] = fmaxf(red[t], red[t + s]);
    __syncthreads();
  }
  mx = red[0]; __syncthreads();
  float sum = 0.f;
  for (int i = 0; i < 16; ++i) { v[i] = __expf(v[i] - mx); sum += v[i]; }
  red[t] = sum; __syncthreads();
  for (int s = 128; s > 0; s >>= 1) {
    if (t < s) red[t] += red[t + s];
    __syncthreads();
  }
  float inv = 1.0f / red[0];
  unsigned short* dst = khat + (size_t)col * SEQ;
  for (int i = 0; i < 16; ++i) dst[t + i * 256] = f2bf(v[i] * inv);
}

// ---------------------------------------------------------------------------
// 2c) v / N, transposed to vhat[b,h,e,n] (bf16) for K-contiguous staging
// ---------------------------------------------------------------------------
__global__ void k_scale_v(const unsigned short* __restrict__ qkvb,
                          unsigned short* __restrict__ vhat) {
  int idx = blockIdx.x * blockDim.x + threadIdx.x;   // over B*N*H*D
  if (idx >= BATCH * SEQ * CH) return;
  int c = idx & 511;            // h*64+e
  int bn = idx >> 9;
  int b = bn >> 12, n = bn & 4095;
  float x = bf2f(qkvb[(size_t)bn * QKVF + 2 * CH + c]);
  vhat[((size_t)b * CH + c) * SEQ + n] = f2bf(x * (1.0f / (float)SEQ));
}

// ---------------------------------------------------------------------------
// 3) context[b,h] = khat(DxN) @ vhat^T(NxD), K=4096. One block per (b,h).
//    8 waves x 2 tiles = 4x4 WMMA tiles. Output transposed: ctxb[b,h,e,d]
// ---------------------------------------------------------------------------
__global__ __launch_bounds__(256) void k_ctx(
    const unsigned short* __restrict__ khat,   // [64][64][4096]
    const unsigned short* __restrict__ vhat,   // [64][64][4096]
    unsigned short* __restrict__ ctxb) {       // [64][64 e][64 d]
  __shared__ alignas(16) unsigned short As[64 * 64];
  __shared__ alignas(16) unsigned short Bs[64 * 64];   // Bt[e][k]
  const int t = threadIdx.x, wave = t >> 5, bh = blockIdx.x;
  const int ti = wave >> 1;
  const int tj0 = (wave & 1) * 2;
  const v8f zero = {0.f, 0.f, 0.f, 0.f, 0.f, 0.f, 0.f, 0.f};
  v8f acc[2] = {zero, zero};
  const unsigned short* Ab = khat + (size_t)bh * DHEAD * SEQ;
  const unsigned short* Bb = vhat + (size_t)bh * DHEAD * SEQ;

  for (int kk = 0; kk < SEQ; kk += 64) {
    for (int c = t; c < 512; c += 256) {          // 64*64/8 vector chunks
      int r = c >> 3, kc = (c & 7) << 3;
      async_copy16(Ab + (size_t)r * SEQ + kk + kc, As + r * 64 + kc);
      async_copy16(Bb + (size_t)r * SEQ + kk + kc, Bs + r * 64 + kc);
    }
    async_wait_all();
    __syncthreads();
    for (int ks = 0; ks < 64; ks += 32) {
      v16bf a  = load_a_frag(As, 64, ti * 16, ks);
      v16bf b0 = load_b_frag(Bs, 64, tj0 * 16, ks);
      v16bf b1 = load_b_frag(Bs, 64, (tj0 + 1) * 16, ks);
      acc[0] = wmma_bf16(a, b0, acc[0]);
      acc[1] = wmma_bf16(a, b1, acc[1]);
    }
    __syncthreads();
  }
  const int lane = t & 31;
  const int cidx = lane & 15, rbase = (lane >> 4) << 3;
  for (int j = 0; j < 2; ++j) {
    int e = (tj0 + j) * 16 + cidx;
    for (int r = 0; r < 8; ++r) {
      int d = ti * 16 + rbase + r;
      ctxb[((size_t)bh * DHEAD + e) * DHEAD + d] = f2bf(acc[j][r]);
    }
  }
}

// ---------------------------------------------------------------------------
// 4) out[b,n,h,e] = qhat(NxD) @ ctx(DxE).  128-row tiles, one (b,h) per blockIdx.y.
//    Wave w: rows w*16..+15, 4 col tiles.  Result -> attnb[b,n,(h e)] bf16.
// ---------------------------------------------------------------------------
__global__ __launch_bounds__(256) void k_apply_q(
    const unsigned short* __restrict__ qhat,   // [64][4096][64]
    const unsigned short* __restrict__ ctxb,   // [64][64 e][64 d]
    unsigned short* __restrict__ attnb) {      // [8][4096][512]
  __shared__ alignas(16) unsigned short As[128 * 64];
  __shared__ alignas(16) unsigned short Bs[64 * 64];   // Bt[e][d]
  const int t = threadIdx.x, wave = t >> 5;
  const int bh = blockIdx.y, b = bh >> 3, h = bh & 7;
  const int row0 = blockIdx.x * 128;
  const unsigned short* Ab = qhat + ((size_t)bh * SEQ + row0) * DHEAD;
  const unsigned short* Bb = ctxb + (size_t)bh * DHEAD * DHEAD;

  for (int c = t; c < 1024; c += 256) {           // 128*64/8
    int r = c >> 3, kc = (c & 7) << 3;
    async_copy16(Ab + (size_t)r * 64 + kc, As + r * 64 + kc);
  }
  for (int c = t; c < 512; c += 256) {            // 64*64/8
    int r = c >> 3, kc = (c & 7) << 3;
    async_copy16(Bb + (size_t)r * 64 + kc, Bs + r * 64 + kc);
  }
  async_wait_all();
  __syncthreads();

  const v8f zero = {0.f, 0.f, 0.f, 0.f, 0.f, 0.f, 0.f, 0.f};
  v8f acc[4] = {zero, zero, zero, zero};
  for (int ks = 0; ks < 64; ks += 32) {
    v16bf a = load_a_frag(As, 64, wave * 16, ks);
    for (int j = 0; j < 4; ++j) {
      v16bf bf = load_b_frag(Bs, 64, j * 16, ks);
      acc[j] = wmma_bf16(a, bf, acc[j]);
    }
  }
  const int lane = t & 31;
  const int cidx = lane & 15, rbase = (lane >> 4) << 3;
  for (int j = 0; j < 4; ++j) {
    int e = j * 16 + cidx;
    for (int r = 0; r < 8; ++r) {
      int n = row0 + wave * 16 + rbase + r;
      attnb[((size_t)b * SEQ + n) * CH + h * DHEAD + e] = f2bf(acc[j][r]);
    }
  }
}

// ---------------------------------------------------------------------------
// 5) out = attn @ w_out + b_out, then LayerNorm(scale) per row -> fp32 d_out.
//    16 rows x 512 cols per block, 8 waves x 4 col tiles, K=512 in 32-chunks.
//    LDS (A|Bt) is reused for the 16x512 fp32 row buffer in the LN epilogue.
// ---------------------------------------------------------------------------
__global__ __launch_bounds__(256) void k_out_ln(
    const unsigned short* __restrict__ attnb,  // [32768][512]
    const unsigned short* __restrict__ woutb,  // [512 f][512 c]
    const float* __restrict__ bias,            // [512]
    const float* __restrict__ lnsc,            // [512]
    float* __restrict__ out) {                 // [32768][512]
  __shared__ alignas(16) unsigned char smemraw[16 * 32 * 2 + 512 * 32 * 2];
  unsigned short* As = (unsigned short*)smemraw;           // 16x32
  unsigned short* Bs = As + 16 * 32;                       // Bt[c][k] 512x32
  __shared__ float redS[16 * 16];
  __shared__ float redQ[16 * 16];
  __shared__ float mrow[16];
  __shared__ float rrow[16];
  const int t = threadIdx.x, wave = t >> 5, lane = t & 31;
  const int row0 = blockIdx.x * 16;

  const v8f zero = {0.f, 0.f, 0.f, 0.f, 0.f, 0.f, 0.f, 0.f};
  v8f acc[4] = {zero, zero, zero, zero};

  for (int kk = 0; kk < CH; kk += 32) {
    if (t < 64) {                                  // A: 16x32 async
      int r = t >> 2, kc = (t & 3) << 3;
      async_copy16(attnb + (size_t)(row0 + r) * CH + kk + kc, As + r * 32 + kc);
    }
    for (int i = 0; i < 64; ++i) {                 // Bt: 512x32 (coalesced reads)
      int e = t + i * 256;                         // 0..16383
      int k = e >> 9, c = e & 511;
      Bs[c * 32 + k] = woutb[(size_t)(kk + k) * CH + c];
    }
    async_wait_all();
    __syncthreads();
    v16bf a = load_a_frag(As, 32, 0, 0);
    for (int j = 0; j < 4; ++j) {
      v16bf bf = load_b_frag(Bs, 32, wave * 64 + j * 16, 0);
      acc[j] = wmma_bf16(a, bf, acc[j]);
    }
    __syncthreads();
  }

  // epilogue: stage rows (with bias) into reused LDS, then LayerNorm
  float* Os = (float*)smemraw;                     // 16 x 512 fp32 (32 KB)
  const int cidx = lane & 15, rbase = (lane >> 4) << 3;
  for (int j = 0; j < 4; ++j) {
    int c = wave * 64 + j * 16 + cidx;
    float bv = bias[c];
    for (int r = 0; r < 8; ++r) Os[(rbase + r) * 512 + c] = acc[j][r] + bv;
  }
  __syncthreads();
  {
    int row = t >> 4, sub = t & 15;
    float s = 0.f, q = 0.f;
    const float* rp = Os + row * 512 + sub * 32;
    for (int i = 0; i < 32; ++i) { float v = rp[i]; s += v; q += v * v; }
    redS[row * 16 + sub] = s;
    redQ[row * 16 + sub] = q;
  }
  __syncthreads();
  if (t < 16) {
    float s = 0.f, q = 0.f;
    for (int i = 0; i < 16; ++i) { s += redS[t * 16 + i]; q += redQ[t * 16 + i]; }
    float mu = s * (1.0f / 512.0f);
    float var = q * (1.0f / 512.0f) - mu * mu;
    mrow[t] = mu;
    rrow[t] = rsqrtf(var + 1e-5f);
  }
  __syncthreads();
  {
    int row = t >> 4, sub = t & 15;
    float mu = mrow[row], rs = rrow[row];
    const float* rp = Os + row * 512;
    float* og = out + (size_t)(row0 + row) * 512;
    for (int i = sub * 32; i < sub * 32 + 32; ++i)
      og[i] = (rp[i] - mu) * rs * lnsc[i];
  }
}

// ---------------------------------------------------------------------------
// launch
// ---------------------------------------------------------------------------
extern "C" void kernel_launch(void* const* d_in, const int* in_sizes, int n_in,
                              void* d_out, int out_size, void* d_ws, size_t ws_size,
                              hipStream_t stream) {
  const float* x        = (const float*)d_in[0];  // (8,4096,512)
  const float* w_qkv    = (const float*)d_in[1];  // (512,1536)
  const float* w_out    = (const float*)d_in[2];  // (512,512)
  const float* b_out    = (const float*)d_in[3];  // (512)
  const float* ln_scale = (const float*)d_in[4];  // (512)
  float* out = (float*)d_out;

  char* ws = (char*)d_ws;
  size_t off = 0;
  auto alloc = [&](size_t bytes) -> void* {
    void* p = ws + off;
    off = (off + bytes + 255) & ~(size_t)255;
    return p;
  };
  const size_t NX  = (size_t)BATCH * SEQ * CH;        // 16,777,216
  const size_t NW1 = (size_t)CH * QKVF;               // 786,432
  const size_t NW2 = (size_t)CH * CH;                 // 262,144
  unsigned short* xb    = (unsigned short*)alloc(NX * 2);
  unsigned short* wqkvb = (unsigned short*)alloc(NW1 * 2);
  unsigned short* woutb = (unsigned short*)alloc(NW2 * 2);
  unsigned short* qkvb  = (unsigned short*)alloc((size_t)ROWS * QKVF * 2);
  unsigned short* qhat  = (unsigned short*)alloc(NX * 2);   // [b,h,n,d]
  unsigned short* khat  = (unsigned short*)alloc(NX * 2);   // [b,h,d,n]
  unsigned short* vhat  = (unsigned short*)alloc(NX * 2);   // [b,h,e,n]
  unsigned short* ctxb  = (unsigned short*)alloc((size_t)64 * 64 * 64 * 2);
  unsigned short* attnb = (unsigned short*)alloc(NX * 2);   // [b,n,(h e)]

  k_f32_to_bf16<<<(int)((NX  + 255) / 256), 256, 0, stream>>>(x, xb, (int)NX);
  k_f32_to_bf16<<<(int)((NW1 + 255) / 256), 256, 0, stream>>>(w_qkv, wqkvb, (int)NW1);
  k_f32_to_bf16<<<(int)((NW2 + 255) / 256), 256, 0, stream>>>(w_out, woutb, (int)NW2);

  k_gemm_qkv<<<dim3(ROWS / 128, QKVF / 128), 256, 0, stream>>>(xb, wqkvb, qkvb);

  k_softmax_q<<<(ROWS * HEADS) / 256, 256, 0, stream>>>(qkvb, qhat);
  k_softmax_k<<<BATCH * HEADS * DHEAD, 256, 0, stream>>>(qkvb, khat);
  k_scale_v<<<(int)(NX / 256), 256, 0, stream>>>(qkvb, vhat);

  k_ctx<<<BATCH * HEADS, 256, 0, stream>>>(khat, vhat, ctxb);
  k_apply_q<<<dim3(SEQ / 128, BATCH * HEADS), 256, 0, stream>>>(qhat, ctxb, attnb);
  k_out_ln<<<ROWS / 16, 256, 0, stream>>>(attnb, woutb, b_out, ln_scale, out);
}
